// GCN_56478819943003
// MI455X (gfx1250) — matrix operand order repaired and checked by reading, not compiled, not run on hardware
//
#include <hip/hip_runtime.h>
#include <hip/hip_bf16.h>
#include <stdint.h>

// ---------------------------------------------------------------------------
// GCN on gfx1250 (MI455X): bf16 WMMA GEMMs + L2-resident fp32 atomic scatter.
//
// Roofline: dense GEMMs ~10 GFLOP (negligible); edge gather/scatter moves
// ~1.4 GB but every activation buffer (max 51 MB) fits the 192 MB L2, so the
// random gathers + global_atomic_add_f32 resolve on-chip. GEMMs use
// v_wmma_f32_16x16x32_bf16 (one wave per 16x16 tile, fp32 accumulate).
// ---------------------------------------------------------------------------

typedef __attribute__((ext_vector_type(16))) __bf16 v16bf;
typedef __attribute__((ext_vector_type(8)))  __bf16 v8bf;
typedef __attribute__((ext_vector_type(8)))  float  v8f;

#define N_GRAPH 64
#define NEG_SLOPE 0.01f

__device__ __forceinline__ __bf16 f2bf(float f) {
    union { float f; unsigned u; } x; x.f = f;
    unsigned r = x.u + 0x7fffu + ((x.u >> 16) & 1u);   // RNE
    union { unsigned short s; __bf16 b; } y; y.s = (unsigned short)(r >> 16);
    return y.b;
}
__device__ __forceinline__ unsigned fenc(float f) {    // order-preserving uint
    unsigned u = __float_as_uint(f);
    return (u & 0x80000000u) ? ~u : (u | 0x80000000u);
}
__device__ __forceinline__ float fdec(unsigned e) {
    unsigned u = (e & 0x80000000u) ? (e & 0x7fffffffu) : ~e;
    return __uint_as_float(u);
}

__global__ __launch_bounds__(256) void k_fill_f32(float* p, float v, int n) {
    int i = blockIdx.x * 256 + threadIdx.x;
    if (i < n) p[i] = v;
}
__global__ __launch_bounds__(256) void k_fill_u32(unsigned* p, unsigned v, int n) {
    int i = blockIdx.x * 256 + threadIdx.x;
    if (i < n) p[i] = v;
}
__global__ __launch_bounds__(256) void k_deg_accum(const int* __restrict__ dst,
                                                   float* __restrict__ deg, int e) {
    int i = blockIdx.x * 256 + threadIdx.x;
    if (i < e) atomicAdd(&deg[dst[i]], 1.0f);
}
__global__ __launch_bounds__(256) void k_rsqrt_inplace(float* d, int n) {
    int i = blockIdx.x * 256 + threadIdx.x;
    if (i < n) d[i] = rsqrtf(d[i]);
}
__global__ __launch_bounds__(256) void k_f32_to_bf16(const float* __restrict__ a,
                                                     __bf16* __restrict__ b, int n) {
    int i = blockIdx.x * 256 + threadIdx.x;
    if (i < n) b[i] = f2bf(a[i]);
}
// W row-major [K][N] -> Wt bf16 [N][K] so WMMA B-columns are K-contiguous.
__global__ __launch_bounds__(256) void k_transpose_bf16(const float* __restrict__ W,
                                                        __bf16* __restrict__ Wt,
                                                        int K, int N) {
    int i = blockIdx.x * 256 + threadIdx.x;
    if (i >= K * N) return;
    int k = i / N, n = i % N;
    Wt[(size_t)n * K + k] = f2bf(W[i]);
}

// C[M,N] = A[M,K](bf16) * Bt[N,K](bf16)^T (+bias)(+relu). One wave per tile.
__global__ __launch_bounds__(256) void k_wmma_gemm_bf16(
    const __bf16* __restrict__ A, const __bf16* __restrict__ Bt,
    float* __restrict__ C, const float* __restrict__ bias,
    int M, int K, int N, int relu)
{
    int lane  = threadIdx.x & 31;
    int wave  = (blockIdx.x * blockDim.x + threadIdx.x) >> 5;
    int tilesN = N >> 4;
    int tiles  = (M >> 4) * tilesN;
    if (wave >= tiles) return;                 // wave-uniform: EXEC stays all-ones

    int row0 = (wave / tilesN) << 4;
    int col0 = (wave % tilesN) << 4;
    int l = lane & 15;
    int g = lane >> 4;

    const __bf16* arow = A  + (size_t)(row0 + l) * K;
    const __bf16* bcol = Bt + (size_t)(col0 + l) * K;

    v8f acc = {};
    for (int k0 = 0; k0 < K; k0 += 32) {
        // A 16x32 bf16: elems 0..7 -> K=k0+g*8+e, elems 8..15 -> K=k0+16+g*8+e
        v8bf a0 = *(const v8bf*)(arow + k0 + g * 8);
        v8bf a1 = *(const v8bf*)(arow + k0 + 16 + g * 8);
        // B 32x16 bf16: elems 0..7 -> K=k0+g*8+e, elems 8..15 -> K=k0+16+g*8+e
        v8bf b0 = *(const v8bf*)(bcol + k0 + g * 8);
        v8bf b1 = *(const v8bf*)(bcol + k0 + 16 + g * 8);
        v16bf av, bv;
#pragma unroll
        for (int e = 0; e < 8; ++e) {
            av[e] = a0[e]; av[e + 8] = a1[e];
            bv[e] = b0[e]; bv[e + 8] = b1[e];
        }
        acc = __builtin_amdgcn_wmma_f32_16x16x32_bf16(
            false, av, false, bv, (short)0, acc, false, false);
    }

    float bb = bias ? bias[col0 + l] : 0.0f;
#pragma unroll
    for (int r = 0; r < 8; ++r) {              // D: VGPR r -> M=g*8+r, N=lane&15
        float v = acc[r] + bb;
        if (relu) v = fmaxf(v, 0.0f);
        C[(size_t)(row0 + g * 8 + r) * N + col0 + l] = v;
    }
}

// out[i,f] = gemm[i,f]*dinv[i]^2 + bias[f]   (self-loop + bias, full overwrite)
__global__ __launch_bounds__(256) void k_scatter_self(
    const float* __restrict__ gemm, const float* __restrict__ dinv,
    const float* __restrict__ bias, float* __restrict__ out,
    int total, int Dmask, int logD)
{
    int i = blockIdx.x * 256 + threadIdx.x;
    if (i >= total) return;
    int node = i >> logD, f = i & Dmask;
    float di = dinv[node];
    out[i] = gemm[i] * di * di + bias[f];
}
// out[dst,f] += gemm[src,f] * dinv[src]*dinv[dst]
__global__ __launch_bounds__(256) void k_scatter_edges(
    const int* __restrict__ src, const int* __restrict__ dst,
    const float* __restrict__ gemm, const float* __restrict__ dinv,
    float* __restrict__ out, long long total, int Dmask, int logD)
{
    long long i = (long long)blockIdx.x * 256 + threadIdx.x;
    if (i >= total) return;
    int e = (int)(i >> logD), f = (int)(i & Dmask);
    int s = src[e], d = dst[e];
    atomicAdd(&out[((size_t)d << logD) | f],
              gemm[((size_t)s << logD) | f] * dinv[s] * dinv[d]);
}
__global__ __launch_bounds__(256) void k_leaky_bf16(const float* __restrict__ in,
                                                    __bf16* __restrict__ out, int n) {
    int i = blockIdx.x * 256 + threadIdx.x;
    if (i >= n) return;
    float v = in[i];
    out[i] = f2bf(v >= 0.0f ? v : NEG_SLOPE * v);
}

__global__ __launch_bounds__(256) void k_segmax(const float* __restrict__ h,
                                                const int* __restrict__ batch,
                                                unsigned* __restrict__ genc, int n) {
    int i = blockIdx.x * 256 + threadIdx.x;
    if (i >= n * 256) return;
    int node = i >> 8, f = i & 255;
    atomicMax(&genc[((unsigned)batch[node] << 8) | f], fenc(h[i]));
}
__global__ __launch_bounds__(256) void k_decode_gmax(const unsigned* __restrict__ genc,
                                                     __bf16* __restrict__ gbf, int n) {
    int i = blockIdx.x * 256 + threadIdx.x;
    if (i < n) gbf[i] = f2bf(fdec(genc[i]));
}

__global__ __launch_bounds__(256) void k_final_fc(const float* __restrict__ g2,
                                                  const float* __restrict__ Wc3,
                                                  const float* __restrict__ bc3,
                                                  float* __restrict__ out) {
    int idx = threadIdx.x;                     // 64 graphs * 4 outputs
    int gi = idx >> 2, o = idx & 3;
    float s = bc3[o];
    for (int k = 0; k < 512; ++k) s += g2[gi * 512 + k] * Wc3[k * 4 + o];
    out[idx] = s;
}

extern "C" void kernel_launch(void* const* d_in, const int* in_sizes, int n_in,
                              void* d_out, int out_size, void* d_ws, size_t ws_size,
                              hipStream_t stream) {
    const float* x     = (const float*)d_in[0];
    const int*   ei    = (const int*)  d_in[1];
    const int*   batch = (const int*)  d_in[2];
    const float* W1 = (const float*)d_in[3];   const float* b1  = (const float*)d_in[4];
    const float* W2 = (const float*)d_in[5];   const float* b2  = (const float*)d_in[6];
    const float* W3 = (const float*)d_in[7];   const float* b3  = (const float*)d_in[8];
    const float* Wc1 = (const float*)d_in[9];  const float* bc1 = (const float*)d_in[10];
    const float* Wc2 = (const float*)d_in[11]; const float* bc2 = (const float*)d_in[12];
    const float* Wc3 = (const float*)d_in[13]; const float* bc3 = (const float*)d_in[14];

    const int N = in_sizes[0] / 128;     // 50000 (multiple of 16)
    const int E = in_sizes[1] / 2;       // 800000
    const int* src = ei;
    const int* dst = ei + E;

    char* ws = (char*)d_ws; size_t off = 0;
    auto alloc = [&](size_t bytes) -> void* {
        void* p = ws + off; off += (bytes + 255) & ~(size_t)255; return p;
    };
    float*  dinv = (float*) alloc((size_t)N * 4);
    __bf16* abf  = (__bf16*)alloc((size_t)N * 128 * 2);
    float*  gemm = (float*) alloc((size_t)N * 256 * 4);
    float*  scat = (float*) alloc((size_t)N * 256 * 4);
    __bf16* W1t  = (__bf16*)alloc(128 * 64 * 2);
    __bf16* W2t  = (__bf16*)alloc(64 * 128 * 2);
    __bf16* W3t  = (__bf16*)alloc(128 * 256 * 2);
    __bf16* Wc1t = (__bf16*)alloc(256 * 1024 * 2);
    __bf16* Wc2t = (__bf16*)alloc(1024 * 512 * 2);
    unsigned* genc = (unsigned*)alloc(N_GRAPH * 256 * 4);
    __bf16* gbf  = (__bf16*)alloc(N_GRAPH * 256 * 2);
    float*  m1   = (float*) alloc(N_GRAPH * 1024 * 4);
    __bf16* m1bf = (__bf16*)alloc(N_GRAPH * 1024 * 2);
    float*  m2   = (float*) alloc(N_GRAPH * 512 * 4);

    auto gr = [](long long n) { return dim3((unsigned)((n + 255) / 256)); };
    auto gemmGrid = [](int M, int Nout) {
        int waves = (M >> 4) * (Nout >> 4);
        return dim3((unsigned)((waves + 7) / 8));
    };

    // degrees with self-loops -> dinv = rsqrt(deg)
    k_fill_f32<<<gr(N), 256, 0, stream>>>(dinv, 1.0f, N);
    k_deg_accum<<<gr(E), 256, 0, stream>>>(dst, dinv, E);
    k_rsqrt_inplace<<<gr(N), 256, 0, stream>>>(dinv, N);

    // bf16 conversions (weights transposed to [N][K])
    k_f32_to_bf16<<<gr((long long)N * 128), 256, 0, stream>>>(x, abf, N * 128);
    k_transpose_bf16<<<gr(128 * 64),   256, 0, stream>>>(W1,  W1t,  128, 64);
    k_transpose_bf16<<<gr(64 * 128),   256, 0, stream>>>(W2,  W2t,  64, 128);
    k_transpose_bf16<<<gr(128 * 256),  256, 0, stream>>>(W3,  W3t,  128, 256);
    k_transpose_bf16<<<gr(256 * 1024), 256, 0, stream>>>(Wc1, Wc1t, 256, 1024);
    k_transpose_bf16<<<gr(1024 * 512), 256, 0, stream>>>(Wc2, Wc2t, 1024, 512);

    // conv1: 128 -> 64, leaky
    k_wmma_gemm_bf16<<<gemmGrid(N, 64), 256, 0, stream>>>(abf, W1t, gemm, nullptr, N, 128, 64, 0);
    k_scatter_self<<<gr((long long)N * 64), 256, 0, stream>>>(gemm, dinv, b1, scat, N * 64, 63, 6);
    k_scatter_edges<<<gr((long long)E * 64), 256, 0, stream>>>(src, dst, gemm, dinv, scat, (long long)E * 64, 63, 6);
    k_leaky_bf16<<<gr((long long)N * 64), 256, 0, stream>>>(scat, abf, N * 64);

    // conv2: 64 -> 128, leaky
    k_wmma_gemm_bf16<<<gemmGrid(N, 128), 256, 0, stream>>>(abf, W2t, gemm, nullptr, N, 64, 128, 0);
    k_scatter_self<<<gr((long long)N * 128), 256, 0, stream>>>(gemm, dinv, b2, scat, N * 128, 127, 7);
    k_scatter_edges<<<gr((long long)E * 128), 256, 0, stream>>>(src, dst, gemm, dinv, scat, (long long)E * 128, 127, 7);
    k_leaky_bf16<<<gr((long long)N * 128), 256, 0, stream>>>(scat, abf, N * 128);

    // conv3: 128 -> 256
    k_wmma_gemm_bf16<<<gemmGrid(N, 256), 256, 0, stream>>>(abf, W3t, gemm, nullptr, N, 128, 256, 0);
    k_scatter_self<<<gr((long long)N * 256), 256, 0, stream>>>(gemm, dinv, b3, scat, N * 256, 255, 8);
    k_scatter_edges<<<gr((long long)E * 256), 256, 0, stream>>>(src, dst, gemm, dinv, scat, (long long)E * 256, 255, 8);

    // segment max pool -> [64, 256]  (0x007FFFFF == order-encoded -inf)
    k_fill_u32<<<gr(N_GRAPH * 256), 256, 0, stream>>>(genc, 0x007FFFFFu, N_GRAPH * 256);
    k_segmax<<<gr((long long)N * 256), 256, 0, stream>>>(scat, batch, genc, N);
    k_decode_gmax<<<gr(N_GRAPH * 256), 256, 0, stream>>>(genc, gbf, N_GRAPH * 256);

    // MLP head: 256 -> 1024 (relu) -> 512 (relu) -> 4
    k_wmma_gemm_bf16<<<gemmGrid(N_GRAPH, 1024), 256, 0, stream>>>(gbf, Wc1t, m1, bc1, N_GRAPH, 256, 1024, 1);
    k_f32_to_bf16<<<gr(N_GRAPH * 1024), 256, 0, stream>>>(m1, m1bf, N_GRAPH * 1024);
    k_wmma_gemm_bf16<<<gemmGrid(N_GRAPH, 512), 256, 0, stream>>>(m1bf, Wc2t, m2, bc2, N_GRAPH, 1024, 512, 1);
    k_final_fc<<<1, 256, 0, stream>>>(m2, Wc3, bc3, (float*)d_out);
}
